// EncDec_23880018165896
// MI455X (gfx1250) — matrix operand order repaired and checked by reading, not compiled, not run on hardware
//
#include <hip/hip_runtime.h>

typedef __attribute__((ext_vector_type(16))) __bf16 v16bf;
typedef __attribute__((ext_vector_type(8)))  float  v8f;

#define KDIM   120000
#define NOUT   423
#define NPAD   432
#define MDIM   64
#define KSPLIT 75
#define KCHUNK 1600      // KDIM / KSPLIT = 1600 = 50 * 32
#define NTILES 27
#define NIMG   192       // 64 * 3
#define IMGW   200
#define IMGPIX 40000

// ---------------------------------------------------------------------------
// GEMM: partials[split][m][n] = sum_{k in split} X[m,k] * W[n,k]  (bf16 WMMA)
// grid = (27 n-tiles, 75 k-splits), block = 128 (4 waves, one 16-row M tile each)
// ---------------------------------------------------------------------------
__global__ __launch_bounds__(128)
void wmma_gemm_kernel(const float* __restrict__ X,
                      const float* __restrict__ W,
                      float* __restrict__ partials)
{
    const int lane = threadIdx.x & 31;
    const int wave = threadIdx.x >> 5;
    const int r    = lane & 15;
    const int half = lane >> 4;
    const int ntile = blockIdx.x;
    const int split = blockIdx.y;

    const int m = wave * 16 + r;            // A: lane holds row m
    int n = ntile * 16 + r;                 // B: lane holds column n (= W row n)
    if (n >= NOUT) n = NOUT - 1;            // clamp for partial last tile

    const float* __restrict__ xrow = X + (size_t)m * KDIM;
    const float* __restrict__ wrow = W + (size_t)n * KDIM;
    const int k0 = split * KCHUNK;

    v8f acc = {};
    #pragma unroll 2
    for (int kk = 0; kk < KCHUNK; kk += 32) {
        // A 16x32 bf16: elems 0..7 -> K = ka..ka+7 ; elems 8..15 -> K = ka+16..ka+23
        const int ka = k0 + kk + half * 8;
        // B 32x16 bf16: elems 0..7 -> K = kb..kb+7 ; elems 8..15 -> K = kb+16..kb+23
        const int kb = k0 + kk + half * 8;

        float4 a0 = *(const float4*)(xrow + ka);
        float4 a1 = *(const float4*)(xrow + ka + 4);
        float4 a2 = *(const float4*)(xrow + ka + 16);
        float4 a3 = *(const float4*)(xrow + ka + 20);
        float4 b0 = *(const float4*)(wrow + kb);
        float4 b1 = *(const float4*)(wrow + kb + 4);
        float4 b2 = *(const float4*)(wrow + kb + 16);
        float4 b3 = *(const float4*)(wrow + kb + 20);

        v16bf A = { (__bf16)a0.x, (__bf16)a0.y, (__bf16)a0.z, (__bf16)a0.w,
                    (__bf16)a1.x, (__bf16)a1.y, (__bf16)a1.z, (__bf16)a1.w,
                    (__bf16)a2.x, (__bf16)a2.y, (__bf16)a2.z, (__bf16)a2.w,
                    (__bf16)a3.x, (__bf16)a3.y, (__bf16)a3.z, (__bf16)a3.w };
        v16bf B = { (__bf16)b0.x, (__bf16)b0.y, (__bf16)b0.z, (__bf16)b0.w,
                    (__bf16)b1.x, (__bf16)b1.y, (__bf16)b1.z, (__bf16)b1.w,
                    (__bf16)b2.x, (__bf16)b2.y, (__bf16)b2.z, (__bf16)b2.w,
                    (__bf16)b3.x, (__bf16)b3.y, (__bf16)b3.z, (__bf16)b3.w };

        acc = __builtin_amdgcn_wmma_f32_16x16x32_bf16(
            false, A, false, B, (short)0, acc, false, false);
    }

    // D layout: VGPR rr -> M = rr + 8*half, N = lane&15
    float* pb = partials + (size_t)split * (MDIM * NPAD);
    const int ncol = ntile * 16 + r;        // padded store, cols 423..431 are junk
    #pragma unroll
    for (int rr = 0; rr < 8; ++rr) {
        const int mrow = wave * 16 + half * 8 + rr;
        pb[mrow * NPAD + ncol] = acc[rr];
    }
}

// ---------------------------------------------------------------------------
// Deterministic split-K reduction + bias; also scatters points_3d / coeffs
// ---------------------------------------------------------------------------
__global__ void reduce_bias_kernel(const float* __restrict__ partials,
                                   const float* __restrict__ b_enc,
                                   float* __restrict__ net,
                                   float* __restrict__ out_p3d,
                                   float* __restrict__ out_coef)
{
    int idx = blockIdx.x * blockDim.x + threadIdx.x;
    if (idx >= MDIM * NOUT) return;
    int m = idx / NOUT, n = idx % NOUT;
    float s = b_enc[n];
    #pragma unroll 5
    for (int sp = 0; sp < KSPLIT; ++sp)
        s += partials[(size_t)sp * (MDIM * NPAD) + m * NPAD + n];
    net[m * NPAD + n] = s;
    if (n < 384) out_p3d[m * 384 + n] = s;
    else         out_coef[m * 39 + (n - 384)] = s;
}

__global__ void fill_zero_kernel(float* __restrict__ p, int n)
{
    int i = blockIdx.x * blockDim.x + threadIdx.x;
    if (i < n) p[i] = 0.0f;
}

// ---------------------------------------------------------------------------
// Camera projection + binary mask scatter (min(count,1) == store 1.0f)
// ---------------------------------------------------------------------------
__global__ void project_scatter_kernel(const float* __restrict__ net,
                                       float* __restrict__ out_p2d,
                                       float* __restrict__ maskA)
{
    int idx = blockIdx.x * blockDim.x + threadIdx.x;
    if (idx >= MDIM * 3 * 128) return;
    int b  = idx / 384;
    int rm = idx % 384;
    int c  = rm / 128;
    int n  = rm % 128;

    const float* nb = net + (size_t)b * NPAD;
    const float* co = nb + 384 + c * 13;
    float p0 = nb[n * 3 + 0], p1 = nb[n * 3 + 1], p2 = nb[n * 3 + 2];

    float pc0 = co[0] * p0 + co[1] * p1 + co[2] * p2 + co[9];
    float pc1 = co[3] * p0 + co[4] * p1 + co[5] * p2 + co[10];
    float pc2 = co[6] * p0 + co[7] * p1 + co[8] * p2 + co[11];

    float z = (fabsf(pc2) > 1e-6f) ? pc2 : 1e-6f;
    float f = co[12];
    float x2 = f * pc0 / z;
    float y2 = f * pc1 / z;

    size_t o = ((size_t)(b * 3 + c) * 128 + n) * 2;
    out_p2d[o]     = x2;
    out_p2d[o + 1] = y2;

    int px = (int)floorf(x2 + 100.0f);
    int py = (int)floorf(y2 + 100.0f);
    px = px < 0 ? 0 : (px > 199 ? 199 : px);
    py = py < 0 ? 0 : (py > 199 ? 199 : py);
    maskA[((size_t)(b * 3 + c) * IMGW + py) * IMGW + px] = 1.0f;
}

// ---------------------------------------------------------------------------
// Separable 7-tap Gaussian blur, reflect padding (sigma = 1)
// ---------------------------------------------------------------------------
__device__ __forceinline__ float gw(int j) {
    const float w[7] = { 0.0044330482f, 0.0540055826f, 0.2420362294f,
                         0.3990502797f,
                         0.2420362294f, 0.0540055826f, 0.0044330482f };
    return w[j];
}

__global__ void blur_v_kernel(const float* __restrict__ in, float* __restrict__ out)
{
    int idx = blockIdx.x * blockDim.x + threadIdx.x;
    if (idx >= NIMG * IMGPIX) return;
    int img = idx / IMGPIX, p = idx % IMGPIX, y = p / IMGW, x = p % IMGW;
    const float* im = in + (size_t)img * IMGPIX;
    float s = 0.0f;
    #pragma unroll
    for (int j = -3; j <= 3; ++j) {
        int yy = y + j;
        yy = yy < 0 ? -yy : (yy > 199 ? 398 - yy : yy);
        s += gw(j + 3) * im[yy * IMGW + x];
    }
    out[idx] = s;
}

__global__ void blur_h_kernel(const float* __restrict__ in, float* __restrict__ out)
{
    int idx = blockIdx.x * blockDim.x + threadIdx.x;
    if (idx >= NIMG * IMGPIX) return;
    int img = idx / IMGPIX, p = idx % IMGPIX, y = p / IMGW, x = p % IMGW;
    const float* im = in + (size_t)img * IMGPIX;
    float s = 0.0f;
    #pragma unroll
    for (int j = -3; j <= 3; ++j) {
        int xx = x + j;
        xx = xx < 0 ? -xx : (xx > 199 ? 398 - xx : xx);
        s += gw(j + 3) * im[y * IMGW + xx];
    }
    out[idx] = s;
}

// ---------------------------------------------------------------------------
// Per-image min/max normalization (one workgroup per image)
// ---------------------------------------------------------------------------
__global__ __launch_bounds__(256)
void normalize_kernel(const float* __restrict__ in, float* __restrict__ out)
{
    __shared__ float smin[256], smax[256];
    const int img = blockIdx.x;
    const float* im = in  + (size_t)img * IMGPIX;
    float*       om = out + (size_t)img * IMGPIX;

    float mn = 3.4e38f, mx = -3.4e38f;
    for (int i = threadIdx.x; i < IMGPIX; i += 256) {
        float v = im[i];
        mn = fminf(mn, v);
        mx = fmaxf(mx, v);
    }
    smin[threadIdx.x] = mn;
    smax[threadIdx.x] = mx;
    __syncthreads();
    for (int s = 128; s > 0; s >>= 1) {
        if (threadIdx.x < s) {
            smin[threadIdx.x] = fminf(smin[threadIdx.x], smin[threadIdx.x + s]);
            smax[threadIdx.x] = fmaxf(smax[threadIdx.x], smax[threadIdx.x + s]);
        }
        __syncthreads();
    }
    float lo = smin[0];
    float rg = smax[0] - lo;
    if (rg > 0.0f) {
        float inv = 1.0f / rg;
        for (int i = threadIdx.x; i < IMGPIX; i += 256)
            om[i] = (im[i] - lo) * inv;
    } else {
        for (int i = threadIdx.x; i < IMGPIX; i += 256)
            om[i] = 0.0f;
    }
}

// ---------------------------------------------------------------------------
extern "C" void kernel_launch(void* const* d_in, const int* in_sizes, int n_in,
                              void* d_out, int out_size, void* d_ws, size_t ws_size,
                              hipStream_t stream)
{
    const float* x     = (const float*)d_in[0];   // (64, 3, 200, 200)
    const float* W_enc = (const float*)d_in[1];   // (423, 120000)
    const float* b_enc = (const float*)d_in[2];   // (423,)

    float* out   = (float*)d_out;
    float* p3d   = out;                 // 64*128*3      = 24576
    float* coef  = out + 24576;         // 64*3*13       = 2496
    float* p2d   = out + 27072;         // 64*3*128*2    = 49152
    float* masks = out + 76224;         // 64*3*200*200  = 7680000

    float* partials = (float*)d_ws;                             // 75*64*432
    float* net      = partials + (size_t)KSPLIT * MDIM * NPAD;  // 64*432
    float* maskA    = net + (size_t)MDIM * NPAD;                // 192*40000

    // 1) GEMM via bf16 WMMA, split-K partials
    dim3 ggrid(NTILES, KSPLIT);
    wmma_gemm_kernel<<<ggrid, 128, 0, stream>>>(x, W_enc, partials);

    // 2) deterministic split-K reduce + bias; emit points_3d / coeffs
    reduce_bias_kernel<<<(MDIM * NOUT + 255) / 256, 256, 0, stream>>>(
        partials, b_enc, net, p3d, coef);

    // 3) binary masks: zero then scatter 1.0
    fill_zero_kernel<<<(NIMG * IMGPIX + 255) / 256, 256, 0, stream>>>(
        maskA, NIMG * IMGPIX);
    project_scatter_kernel<<<(MDIM * 3 * 128 + 255) / 256, 256, 0, stream>>>(
        net, p2d, maskA);

    // 4) separable Gaussian blur (ping-pong through d_out masks region)
    blur_v_kernel<<<(NIMG * IMGPIX + 255) / 256, 256, 0, stream>>>(maskA, masks);
    blur_h_kernel<<<(NIMG * IMGPIX + 255) / 256, 256, 0, stream>>>(masks, maskA);

    // 5) per-image min/max normalize -> final masks in d_out
    normalize_kernel<<<NIMG, 256, 0, stream>>>(maskA, masks);
}
